// CIFAR_SSM_Classifier_15530601742693
// MI455X (gfx1250) — matrix operand order repaired and checked by reading, not compiled, not run on hardware
//
#include <hip/hip_runtime.h>
#include <hip/hip_bf16.h>

// ---------------------------------------------------------------------------
// CDNA5 (gfx1250) implementation of the CIFAR-SSM classifier.
// Heavy convs (conv2/conv3, ~232 GFLOP of the ~237 total) run as implicit
// GEMM on v_wmma_f32_16x16x32_bf16 (wave32). SSM input/output projections are
// also WMMA GEMMs. Everything else is cheap VALU work.
// ---------------------------------------------------------------------------

typedef __attribute__((ext_vector_type(16))) __bf16          v16bf;
typedef __attribute__((ext_vector_type(8)))  float           v8f;
typedef __attribute__((ext_vector_type(16))) unsigned short  v16u;
typedef __attribute__((ext_vector_type(8)))  unsigned short  v8u;

#define BN_EPS 1e-5f

static __device__ __forceinline__ unsigned short f2bf(float f) {
  unsigned u = __builtin_bit_cast(unsigned, f);
  unsigned r = u + 0x7FFFu + ((u >> 16) & 1u);            // round-to-nearest-even
  if ((u & 0x7F800000u) == 0x7F800000u) r = u;            // inf/nan: truncate
  return (unsigned short)(r >> 16);
}

static __device__ __forceinline__ v16bf as_bf(v16u u) {
  return __builtin_bit_cast(v16bf, u);
}

static __device__ __forceinline__ v8f vzero8() {
  v8f z;
#pragma unroll
  for (int i = 0; i < 8; ++i) z[i] = 0.f;
  return z;
}

// ---------------------------------------------------------------------------
// conv1: (512,3,32,32) NCHW f32 -> (512,32,32,64) NHWC bf16, BN+ReLU folded.
// K = 27: too small for WMMA, direct VALU (1.8 GFLOP).
// ---------------------------------------------------------------------------
__global__ __launch_bounds__(256) void conv1_kernel(
    const float* __restrict__ x, const float* __restrict__ w,
    const float* __restrict__ g, const float* __restrict__ bb,
    const float* __restrict__ mm, const float* __restrict__ vv,
    unsigned short* __restrict__ z1) {
  int idx = blockIdx.x * 256 + threadIdx.x;      // 512*32*32*64 threads
  int oc = idx & 63;
  int p  = idx >> 6;
  int xx = p & 31;
  int yy = (p >> 5) & 31;
  int b  = p >> 10;
  float acc = 0.f;
#pragma unroll
  for (int ic = 0; ic < 3; ++ic)
#pragma unroll
    for (int ky = 0; ky < 3; ++ky) {
      int iy = yy + ky - 1;
      if (iy < 0 || iy > 31) continue;
#pragma unroll
      for (int kx = 0; kx < 3; ++kx) {
        int ix = xx + kx - 1;
        if (ix < 0 || ix > 31) continue;
        acc += x[((b * 3 + ic) * 32 + iy) * 32 + ix] *
               w[((oc * 3 + ic) * 3 + ky) * 3 + kx];
      }
    }
  float inv = g[oc] * rsqrtf(vv[oc] + BN_EPS);
  float out = fmaxf(acc * inv + (bb[oc] - mm[oc] * inv), 0.f);
  z1[idx] = f2bf(out);
}

// ---------------------------------------------------------------------------
// Pack conv weights (OIHW f32, BN-scaled) into B-fragment-ready bf16 layout:
//   wpack[t][ic/32][oc/16][lane][16]  with  B[k][n]: k = hi*8+(e&7)+(e>>3)*16, n = lo
// Also emits per-oc bias = beta - mean*inv.
// ---------------------------------------------------------------------------
__global__ __launch_bounds__(256) void pack_conv_kernel(
    const float* __restrict__ w, const float* __restrict__ g,
    const float* __restrict__ bb, const float* __restrict__ mm,
    const float* __restrict__ vv, unsigned short* __restrict__ wpack,
    float* __restrict__ bias, int IC, int OC) {
  int idx = blockIdx.x * 256 + threadIdx.x;
  int total = IC * OC * 9;
  if (idx < total) {
    int e = idx & 15;
    int lane = (idx >> 4) & 31;
    int rest = idx >> 9;
    int OCB = OC >> 4, KB = IC >> 5;
    int ocb = rest % OCB; rest /= OCB;
    int kb = rest % KB;
    int t = rest / KB;
    int lo = lane & 15, hi = lane >> 4;
    int oc = ocb * 16 + lo;
    int ic = kb * 32 + hi * 8 + (e & 7) + ((e >> 3) << 4);
    float inv = g[oc] * rsqrtf(vv[oc] + BN_EPS);
    wpack[idx] = f2bf(w[(oc * IC + ic) * 9 + t] * inv);
  }
  if (idx < OC) {
    float inv = g[idx] * rsqrtf(vv[idx] + BN_EPS);
    bias[idx] = bb[idx] - mm[idx] * inv;
  }
}

// Pack a row-major [N][K] f32 matrix into B-fragment layout bf16.
__global__ __launch_bounds__(256) void pack_mat_kernel(
    const float* __restrict__ mat, unsigned short* __restrict__ pk,
    int N, int K) {
  int idx = blockIdx.x * 256 + threadIdx.x;
  if (idx >= N * K) return;
  int e = idx & 15;
  int lane = (idx >> 4) & 31;
  int rest = idx >> 9;
  int NBLK = N >> 4;
  int nblk = rest % NBLK;
  int kblk = rest / NBLK;
  int lo = lane & 15, hi = lane >> 4;
  pk[idx] = f2bf(mat[(nblk * 16 + lo) * K + kblk * 32 + hi * 8 + (e & 7) + ((e >> 3) << 4)]);
}

// ---------------------------------------------------------------------------
// 3x3 SAME conv as implicit GEMM on WMMA bf16.
// One block = 8 wave32 waves = tile of 64 pixels (2 rows x 32) x 128 OC.
// LDS stages a 4x34xIC patch (zero halo, padded pitch vs bank conflicts).
// MEAN=false: write NHWC bf16 (+BN bias, ReLU).
// MEAN=true : fused width-mean -> atomicAdd into out_mean[b][y][oc] (f32).
// ---------------------------------------------------------------------------
template <int IC, bool MEAN>
__global__ __launch_bounds__(256) void conv3x3_wmma_kernel(
    const unsigned short* __restrict__ in,     // [B][32][32][IC] bf16
    const unsigned short* __restrict__ wpack,  // [9][IC/32][8][32][16]
    const float* __restrict__ bias,            // [128]
    unsigned short* __restrict__ out_bf,       // [B][32][32][128]
    float* __restrict__ out_mean) {            // [B][32][128]
  constexpr int KB = IC / 32;
  constexpr int LDSP = IC + 8;                 // padded per-pixel pitch (halfs)
  __shared__ unsigned short tile[4 * 34 * LDSP];

  const int tid = threadIdx.x;
  const int b = blockIdx.x >> 4;
  const int ybase = (blockIdx.x & 15) * 2;

  {  // zero entire patch (covers halo)
    unsigned* t32 = (unsigned*)tile;
    const int nd = (4 * 34 * LDSP) / 2;
    for (int i = tid; i < nd; i += 256) t32[i] = 0u;
  }
  __syncthreads();
  // load valid input rows ybase-1..ybase+2 into cols 1..32
  for (int r = 0; r < 4; ++r) {
    int y = ybase - 1 + r;
    if (y < 0 || y > 31) continue;
    const v8u* src = (const v8u*)(in + ((size_t)(b * 32 + y) * 32) * IC);
    const int nch = (32 * IC) / 8;
    for (int i = tid; i < nch; i += 256) {
      int pix = i / (IC / 8);
      int sub = i % (IC / 8);
      *(v8u*)&tile[(r * 34 + 1 + pix) * LDSP + sub * 8] = src[i];
    }
  }
  __syncthreads();

  const int wave = tid >> 5, lane = tid & 31;
  const int lo = lane & 15, hi = lane >> 4;
  const int pt = wave & 3, og = wave >> 2;
  const int m = pt * 16 + lo;          // A-matrix row (pixel) for this lane
  const int py = m >> 5;               // row within the 2-row block
  const int px = m & 31;

  v8f acc[4] = {vzero8(), vzero8(), vzero8(), vzero8()};

  for (int ky = 0; ky < 3; ++ky) {
    for (int kx = 0; kx < 3; ++kx) {
      const int rl = py + ky;
      const int cl = px + kx;
      const int t = ky * 3 + kx;
#pragma unroll
      for (int kb = 0; kb < KB; ++kb) {
        // A fragment: K = hi*8 + e (e<8), 16 + hi*8 + (e-8) (e>=8)
        const int abase = (rl * 34 + cl) * LDSP + kb * 32 + hi * 8;
        v8u a0 = *(const v8u*)&tile[abase];
        v8u a1 = *(const v8u*)&tile[abase + 16];
        v16u av;
#pragma unroll
        for (int i = 0; i < 8; ++i) { av[i] = a0[i]; av[i + 8] = a1[i]; }
        const unsigned short* wp = wpack + (size_t)(t * KB + kb) * 8 * 32 * 16;
#pragma unroll
        for (int j = 0; j < 4; ++j) {
          v16u bv = *(const v16u*)&wp[((og * 4 + j) * 32 + lane) * 16];
          acc[j] = __builtin_amdgcn_wmma_f32_16x16x32_bf16(
              false, as_bf(av), false, as_bf(bv), (short)0, acc[j], false, false);
        }
      }
    }
  }

#pragma unroll
  for (int j = 0; j < 4; ++j) {
    const int n = (og * 4 + j) * 16 + lo;   // D: N = lo
    const float bz = bias[n];
    if (!MEAN) {
#pragma unroll
      for (int r = 0; r < 8; ++r) {          // D: M = r + hi*8
        int mp = pt * 16 + hi * 8 + r;
        int y = ybase + (mp >> 5);
        int xx = mp & 31;
        float val = fmaxf(acc[j][r] + bz, 0.f);
        out_bf[(((size_t)(b * 32 + y) * 32 + xx) * 128) + n] = f2bf(val);
      }
    } else {
      float s = 0.f;
#pragma unroll
      for (int r = 0; r < 8; ++r) s += fmaxf(acc[j][r] + bz, 0.f);
      s += __shfl_xor(s, 16, 32);            // fold hi/lo half-waves
      if (hi == 0) {
        int y = ybase + (pt >> 1);
        atomicAdd(&out_mean[((size_t)b * 32 + y) * 128 + n], s * (1.f / 32.f));
      }
    }
  }
}

// ---------------------------------------------------------------------------
// Generic bf16 WMMA GEMM: C[M][N] (f32) = A[M][K] (bf16) x Bpack (+C if acc).
// Block tile 64x128, grid = (M/64, N/128).
// ---------------------------------------------------------------------------
__global__ __launch_bounds__(256) void gemm_wmma_kernel(
    const unsigned short* __restrict__ A,   // [M][K] bf16 row-major
    const unsigned short* __restrict__ Bp,  // [K/32][N/16][32][16]
    float* __restrict__ C, int M, int K, int N, int accumulate) {
  const int mbase = blockIdx.x * 64;
  const int nbase = blockIdx.y * 128;
  const int NBLK = N >> 4, KB = K >> 5;
  const int tid = threadIdx.x;
  const int wave = tid >> 5, lane = tid & 31;
  const int lo = lane & 15, hi = lane >> 4;
  const int pt = wave & 3, og = wave >> 2;
  const int row = mbase + pt * 16 + lo;

  v8f acc[4] = {vzero8(), vzero8(), vzero8(), vzero8()};
  for (int kb = 0; kb < KB; ++kb) {
    const unsigned short* ap = A + (size_t)row * K + kb * 32 + hi * 8;
    v8u a0 = *(const v8u*)ap;
    v8u a1 = *(const v8u*)(ap + 16);
    v16u av;
#pragma unroll
    for (int i = 0; i < 8; ++i) { av[i] = a0[i]; av[i + 8] = a1[i]; }
#pragma unroll
    for (int j = 0; j < 4; ++j) {
      const int nblk = (nbase >> 4) + og * 4 + j;
      v16u bv = *(const v16u*)&Bp[(((size_t)kb * NBLK + nblk) * 32 + lane) * 16];
      acc[j] = __builtin_amdgcn_wmma_f32_16x16x32_bf16(
          false, as_bf(av), false, as_bf(bv), (short)0, acc[j], false, false);
    }
  }
#pragma unroll
  for (int j = 0; j < 4; ++j) {
    const int n = nbase + (og * 4 + j) * 16 + lo;
#pragma unroll
    for (int r = 0; r < 8; ++r) {
      int mrow = mbase + pt * 16 + hi * 8 + r;
      size_t o = (size_t)mrow * N + n;
      float v = acc[j][r];
      if (accumulate) v += C[o];
      C[o] = v;
    }
  }
}

// u (f32, [B][32][128]) -> u_bf (bf16) and ubar_bf (mean over L, bf16).
__global__ __launch_bounds__(256) void upack_kernel(
    const float* __restrict__ u, unsigned short* __restrict__ u_bf,
    unsigned short* __restrict__ ubar_bf) {
  int idx = blockIdx.x * 256 + threadIdx.x;   // 512*128
  int i = idx & 127, b = idx >> 7;
  float s = 0.f;
  for (int l = 0; l < 32; ++l) {
    float val = u[((size_t)(b * 32 + l)) * 128 + i];
    s += val;
    u_bf[((size_t)(b * 32 + l)) * 128 + i] = f2bf(val);
  }
  ubar_bf[idx] = f2bf(s * (1.f / 32.f));
}

// Diagonal SSM scan over L=32; emits xbar = mean_l(x_l) as bf16.
__global__ __launch_bounds__(256) void ssm_scan_kernel(
    const float* __restrict__ Bu, const float* __restrict__ ssmA,
    const float* __restrict__ h0, unsigned short* __restrict__ xbar_bf) {
  int idx = blockIdx.x * 256 + threadIdx.x;   // 512*256
  int s = idx & 255, b = idx >> 8;
  float a = ssmA[s];
  float ad = -(fmaxf(a, 0.f) + log1pf(expf(-fabsf(a))));   // -softplus(a)
  float h = h0[s], sum = 0.f;
  for (int l = 0; l < 32; ++l) {
    h = h * ad + Bu[((size_t)(b * 32 + l)) * 256 + s];
    sum += h;
  }
  xbar_bf[idx] = f2bf(sum * (1.f / 32.f));
}

__global__ __launch_bounds__(256) void head1_kernel(
    const float* __restrict__ pooled, const float* __restrict__ w1,
    const float* __restrict__ b1, float* __restrict__ h) {
  int idx = blockIdx.x * 256 + threadIdx.x;   // 512*128
  int j = idx & 127, b = idx >> 7;
  float acc = b1[j];
  for (int o = 0; o < 256; ++o) acc += pooled[b * 256 + o] * w1[j * 256 + o];
  h[idx] = fmaxf(acc, 0.f);
}

__global__ __launch_bounds__(256) void head2_kernel(
    const float* __restrict__ h, const float* __restrict__ w2,
    const float* __restrict__ b2, float* __restrict__ out) {
  int idx = blockIdx.x * 256 + threadIdx.x;   // 512*10
  if (idx >= 5120) return;
  int k = idx % 10, b = idx / 10;
  float acc = b2[k];
  for (int j = 0; j < 128; ++j) acc += h[b * 128 + j] * w2[k * 128 + j];
  out[idx] = acc;
}

// ---------------------------------------------------------------------------
extern "C" void kernel_launch(void* const* d_in, const int* in_sizes, int n_in,
                              void* d_out, int out_size, void* d_ws, size_t ws_size,
                              hipStream_t stream) {
  (void)in_sizes; (void)n_in; (void)out_size; (void)ws_size;
  const float* x       = (const float*)d_in[0];
  const float* conv1_w = (const float*)d_in[1];
  const float* conv2_w = (const float*)d_in[2];
  const float* conv3_w = (const float*)d_in[3];
  const float* bn1_g = (const float*)d_in[4],  *bn1_b = (const float*)d_in[5];
  const float* bn1_m = (const float*)d_in[6],  *bn1_v = (const float*)d_in[7];
  const float* bn2_g = (const float*)d_in[8],  *bn2_b = (const float*)d_in[9];
  const float* bn2_m = (const float*)d_in[10], *bn2_v = (const float*)d_in[11];
  const float* bn3_g = (const float*)d_in[12], *bn3_b = (const float*)d_in[13];
  const float* bn3_m = (const float*)d_in[14], *bn3_v = (const float*)d_in[15];
  const float* ssm_A = (const float*)d_in[16];
  const float* ssm_B = (const float*)d_in[17];
  const float* ssm_C = (const float*)d_in[18];
  const float* ssm_D = (const float*)d_in[19];
  const float* ssm_h0 = (const float*)d_in[20];
  const float* head_w1 = (const float*)d_in[21], *head_b1 = (const float*)d_in[22];
  const float* head_w2 = (const float*)d_in[23], *head_b2 = (const float*)d_in[24];

  float* out_logits = (float*)d_out;            // (512,10)
  float* out_act    = (float*)d_out + 5120;     // (512,256) == pooled

  char* ws = (char*)d_ws;
  size_t off = 0;
  auto alloc = [&](size_t bytes) -> char* {
    char* p = ws + off;
    off += (bytes + 255) & ~(size_t)255;
    return p;
  };
  unsigned short* z1      = (unsigned short*)alloc((size_t)512 * 1024 * 64 * 2);
  unsigned short* z2      = (unsigned short*)alloc((size_t)512 * 1024 * 128 * 2);
  unsigned short* wpack2  = (unsigned short*)alloc((size_t)73728 * 2);
  float*          bias2   = (float*)alloc(128 * 4);
  unsigned short* wpack3  = (unsigned short*)alloc((size_t)147456 * 2);
  float*          bias3   = (float*)alloc(128 * 4);
  float*          u       = (float*)alloc((size_t)512 * 32 * 128 * 4);
  unsigned short* u_bf    = (unsigned short*)alloc((size_t)512 * 32 * 128 * 2);
  unsigned short* ubar_bf = (unsigned short*)alloc((size_t)512 * 128 * 2);
  unsigned short* Bpack   = (unsigned short*)alloc((size_t)32768 * 2);
  float*          Bu      = (float*)alloc((size_t)32 * 512 * 256 * 4);
  unsigned short* Cpack   = (unsigned short*)alloc((size_t)65536 * 2);
  unsigned short* Dpack   = (unsigned short*)alloc((size_t)32768 * 2);
  unsigned short* xbar_bf = (unsigned short*)alloc((size_t)512 * 256 * 2);
  float*          hbuf    = (float*)alloc((size_t)512 * 128 * 4);

  // conv3 accumulates its width-mean with atomics: zero u every call.
  hipMemsetAsync(u, 0, (size_t)512 * 32 * 128 * 4, stream);

  // weight / operand packing (tiny)
  pack_conv_kernel<<<288, 256, 0, stream>>>(conv2_w, bn2_g, bn2_b, bn2_m, bn2_v,
                                            wpack2, bias2, 64, 128);
  pack_conv_kernel<<<576, 256, 0, stream>>>(conv3_w, bn3_g, bn3_b, bn3_m, bn3_v,
                                            wpack3, bias3, 128, 128);
  pack_mat_kernel<<<128, 256, 0, stream>>>(ssm_B, Bpack, 256, 128);
  pack_mat_kernel<<<256, 256, 0, stream>>>(ssm_C, Cpack, 256, 256);
  pack_mat_kernel<<<128, 256, 0, stream>>>(ssm_D, Dpack, 256, 128);

  // stem
  conv1_kernel<<<131072, 256, 0, stream>>>(x, conv1_w, bn1_g, bn1_b, bn1_m, bn1_v, z1);
  conv3x3_wmma_kernel<64, false><<<8192, 256, 0, stream>>>(z1, wpack2, bias2, z2, nullptr);
  conv3x3_wmma_kernel<128, true><<<8192, 256, 0, stream>>>(z2, wpack3, bias3, nullptr, u);

  // SSM
  upack_kernel<<<256, 256, 0, stream>>>(u, u_bf, ubar_bf);
  gemm_wmma_kernel<<<dim3(256, 2), 256, 0, stream>>>(u_bf, Bpack, Bu, 16384, 128, 256, 0);
  ssm_scan_kernel<<<512, 256, 0, stream>>>(Bu, ssm_A, ssm_h0, xbar_bf);
  gemm_wmma_kernel<<<dim3(8, 2), 256, 0, stream>>>(xbar_bf, Cpack, out_act, 512, 256, 256, 0);
  gemm_wmma_kernel<<<dim3(8, 2), 256, 0, stream>>>(ubar_bf, Dpack, out_act, 512, 128, 256, 1);

  // head
  head1_kernel<<<256, 256, 0, stream>>>(out_act, head_w1, head_b1, hbuf);
  head2_kernel<<<20, 256, 0, stream>>>(hbuf, head_w2, head_b2, out_logits);
}
